// NeuralJumpSDE_62852551409965
// MI455X (gfx1250) — matrix operand order repaired and checked by reading, not compiled, not run on hardware
//
#include <hip/hip_runtime.h>
#include <hip/hip_bf16.h>

typedef __attribute__((ext_vector_type(16))) _Float16 v16h;
typedef __attribute__((ext_vector_type(8)))  _Float16 v8h;
typedef __attribute__((ext_vector_type(8)))  float    v8f;

#define BATCH    65536
#define HORIZON  128
#define HID      128
#define TILE_M   128
#define NTHREADS 256
#define HSTR     136   // f16 stride for H / hidden weights (272B: 16B aligned, bank-skewed)
#define ISTR     40    // f16 stride for INP / layer-0 weights (80B)
#define LN_EPS   1e-5f

// ---------------- LDS budget (fits 320KB/WGP on CDNA5) ----------------
constexpr int WH_E   = 3*2*HID*HSTR;   // hidden weights W1,W2 (f16, [n][k])  = 104448
constexpr int W0T_E  = 3*HID*ISTR;     // layer-0 weights, K padded 11->32    =  15360
constexpr int WLT_E  = 3*2*HID;        // head weights [o][k]                 =    768
constexpr int INP_E  = TILE_M*ISTR;    // padded input [x|cond|0...]          =   5120
constexpr int H_E    = TILE_M*HSTR;    // activation buffer                   =  17408
constexpr int F16_E  = WH_E + W0T_E + WLT_E + INP_E + H_E;                 // 143104 f16
constexpr int B012_E = 3*3*HID;        // biases b0,b1,b2 (f32)
constexpr int BL_E   = 8;              // head biases
constexpr int G_E    = 3*2*HID;        // LN gains
constexpr int BE_E   = 3*2*HID;        // LN shifts
constexpr int OUTS_E = 3*TILE_M*2;     // per-row MLP outputs
constexpr int RED_E  = 2*NTHREADS;     // LN partial sums
constexpr int XC_E   = TILE_M;         // current state x
constexpr size_t SHMEM_BYTES =
    (size_t)F16_E*2 + (size_t)(B012_E+BL_E+G_E+BE_E+OUTS_E+RED_E+XC_E)*4;  // 302624 B

struct MLPPtrs {
  const float* W[4];
  const float* b[4];
  const float* g[2];
  const float* e[2];
};
struct Params { MLPPtrs mlp[3]; };  // 0=diff, 1=intensity, 2=jump (sorted keys)

// ---------------- math helpers ----------------
__device__ __forceinline__ float softplusf(float z) {
  float r = log1pf(expf(-fabsf(z)));
  return z > 0.f ? z + r : r;
}
__device__ __forceinline__ float gelu_exact(float z) {
  return 0.5f * z * (1.f + erff(z * 0.7071067811865475f));
}
__device__ __forceinline__ float act_apply(float v, int sp) {
  return sp ? softplusf(v) : gelu_exact(v);
}

// ---------------- WMMA fragment loads (CDNA5 16x16x32 f16 layouts) ----------------
// A (16x32, MxK): lanes 0-15 row M=lane hold K 0-15; lanes 16-31 hold K 16-31.
__device__ __forceinline__ v16h load_a_frag(const _Float16* rowbase, int stride, int lane, int kb) {
  const int m  = lane & 15;
  const int hi = lane >> 4;
  const _Float16* p = rowbase + m*stride + kb*32;
  v8h c0 = *(const v8h*)(p + (hi ? 16 : 0));
  v8h c1 = *(const v8h*)(p + (hi ? 24 : 8));
  return __builtin_shufflevector(c0, c1, 0,1,2,3,4,5,6,7,8,9,10,11,12,13,14,15);
}
// B (32x16, KxN) from W^T stored [n][k]: lanes 0-15 (N=lane) hold K 0-15; lanes 16-31 hold K 16-31.
__device__ __forceinline__ v16h load_b_frag(const _Float16* wT, int stride, int lane, int nt, int kb) {
  const int n  = (nt << 4) + (lane & 15);
  const int hi = lane >> 4;
  const _Float16* p = wT + n*stride + kb*32 + (hi ? 16 : 0);
  v8h c0 = *(const v8h*)(p);
  v8h c1 = *(const v8h*)(p + 8);
  return __builtin_shufflevector(c0, c1, 0,1,2,3,4,5,6,7,8,9,10,11,12,13,14,15);
}
// D (16x16 f32): lanes 0-15 N=lane, VGPR r -> M=r; lanes 16-31 -> M=8+r.  Bias+act, write f16.
__device__ __forceinline__ void store_act(_Float16* H, const v8f* acc, const float* bias,
                                          int sp, int wv, int lane) {
  const int col0 = lane & 15;
  const int hi   = lane >> 4;
  const int wrow = wv*16 + hi*8;
#pragma unroll
  for (int nt = 0; nt < 8; ++nt) {
    const int col = nt*16 + col0;
    const float bb = bias[col];
#pragma unroll
    for (int r = 0; r < 8; ++r) {
      float v = acc[nt][r] + bb;
      H[(wrow + r)*HSTR + col] = (_Float16)act_apply(v, sp);
    }
  }
}

// ---------------- kernel ----------------
__global__ void __launch_bounds__(NTHREADS)
neural_jump_sde_kernel(Params P,
                       const float* __restrict__ x0,
                       const float* __restrict__ cond,
                       const float* __restrict__ uni,
                       const float* __restrict__ jnz,
                       const float* __restrict__ dnz,
                       float* __restrict__ out) {
  extern __shared__ char smem_raw[];
  _Float16* WH  = (_Float16*)smem_raw;   // [3][2][128][HSTR]
  _Float16* W0T = WH  + WH_E;            // [3][128][ISTR]
  _Float16* WLT = W0T + W0T_E;           // [3][2][128]
  _Float16* INP = WLT + WLT_E;           // [128][ISTR]
  _Float16* H   = INP + INP_E;           // [128][HSTR]
  float* B012 = (float*)(H + H_E);       // [3][3][128]
  float* BL   = B012 + B012_E;           // [3][2]
  float* G    = BL   + BL_E;             // [3][2][128]
  float* BE   = G    + G_E;
  float* OUTS = BE   + BE_E;             // [3][128][2]
  float* REDS = OUTS + OUTS_E;           // [256]
  float* REDQ = REDS + NTHREADS;         // [256]
  float* XC   = REDQ + NTHREADS;         // [128]

  const int t       = threadIdx.x;
  const int lane    = t & 31;
  const int wv      = t >> 5;
  const int rowbase = blockIdx.x * TILE_M;

  // ======== one-time weight staging: f32 global -> transposed bank-padded f16 LDS ========
  for (int idx = t; idx < 3*2*HID*HID; idx += NTHREADS) {   // hidden W1,W2
    int m = idx >> 15, rem = idx & 32767;
    int l = rem >> 14, e = rem & 16383;
    int k = e >> 7,    n = e & 127;                          // global read coalesced over n
    WH[((m*2 + l)*HID + n)*HSTR + k] = (_Float16)P.mlp[m].W[l+1][k*HID + n];
  }
  for (int idx = t; idx < 3*HID*ISTR; idx += NTHREADS) {    // layer-0, K padded to 32(+bank pad)
    int m = idx / (HID*ISTR), rem = idx % (HID*ISTR);
    int n = rem / ISTR, k = rem % ISTR;
    float v = (k < 11) ? P.mlp[m].W[0][k*HID + n] : 0.f;
    W0T[(m*HID + n)*ISTR + k] = (_Float16)v;
  }
  for (int idx = t; idx < 3*2*HID; idx += NTHREADS) {       // head weights [o][k]
    int m = idx >> 8, o = (idx >> 7) & 1, k = idx & 127;
    int od = (m == 1) ? 1 : 2;
    float v = (o < od) ? P.mlp[m].W[3][k*od + o] : 0.f;
    WLT[idx] = (_Float16)v;
  }
  for (int idx = t; idx < 3*3*HID; idx += NTHREADS) {       // biases b0..b2
    int m = idx / 384, rem = idx % 384;
    int l = rem >> 7, j = rem & 127;
    B012[idx] = P.mlp[m].b[l][j];
  }
  if (t < 6) {                                              // head biases
    int m = t >> 1, o = t & 1;
    int od = (m == 1) ? 1 : 2;
    BL[t] = (o < od) ? P.mlp[m].b[3][o] : 0.f;
  }
  for (int idx = t; idx < 3*2*HID; idx += NTHREADS) {       // LN params
    int m = idx >> 8, l = (idx >> 7) & 1, j = idx & 127;
    G[idx]  = P.mlp[m].g[l][j];
    BE[idx] = P.mlp[m].e[l][j];
  }
  // static input columns: col0=x (per step), cols 1..10 = conditioning, 11..39 = 0
  for (int idx = t; idx < TILE_M*ISTR; idx += NTHREADS) {
    int r = idx / ISTR, c = idx % ISTR;
    float v = 0.f;
    if (c >= 1 && c <= 10) v = cond[(long)(rowbase + r)*10 + (c - 1)];
    INP[idx] = (_Float16)v;
  }
  if (t < TILE_M) {
    float xv = x0[rowbase + t];
    XC[t] = xv;
    out[(long)(rowbase + t)*(HORIZON + 1)] = xv;             // paths[:,0,:]
  }
  __syncthreads();

  // ======== time loop: all weights LDS-resident, only noise streamed from HBM ========
  for (int s = 0; s < HORIZON; ++s) {
    if (t < TILE_M) INP[t*ISTR] = (_Float16)XC[t];
    __syncthreads();

    for (int m = 0; m < 3; ++m) {
      const int sp = (m == 1);                               // intensity MLP uses softplus act

      // ---- layer 0: [128 x 32(pad)] @ [32 x 128], single K block ----
      {
        const _Float16* Wm   = W0T + m*HID*ISTR;
        const float*    bias = B012 + (m*3 + 0)*HID;
        v16h a = load_a_frag(INP + wv*16*ISTR, ISTR, lane, 0);
        v8f acc[8];
#pragma unroll
        for (int nt = 0; nt < 8; ++nt) {
          v8f c = {};
          v16h b = load_b_frag(Wm, ISTR, lane, nt, 0);
          acc[nt] = __builtin_amdgcn_wmma_f32_16x16x32_f16(
              false, a, false, b, (short)0, c, false, false);
        }
        store_act(H, acc, bias, sp, wv, lane);
      }
      __syncthreads();

      // ---- hidden layers 1,2: [128x128]@[128x128] in-place, then LayerNorm ----
      for (int l = 0; l < 2; ++l) {
        const _Float16* Wm   = WH + (m*2 + l)*HID*HSTR;
        const float*    bias = B012 + (m*3 + 1 + l)*HID;
        v16h a[4];
#pragma unroll
        for (int kb = 0; kb < 4; ++kb)
          a[kb] = load_a_frag(H + wv*16*HSTR, HSTR, lane, kb);
        v8f acc[8];
#pragma unroll
        for (int nt = 0; nt < 8; ++nt) {
          v8f c = {};
#pragma unroll
          for (int kb = 0; kb < 4; ++kb) {
            v16h b = load_b_frag(Wm, HSTR, lane, nt, kb);
            c = __builtin_amdgcn_wmma_f32_16x16x32_f16(
                false, a[kb], false, b, (short)0, c, false, false);
          }
          acc[nt] = c;
        }
        store_act(H, acc, bias, sp, wv, lane);   // in-place: wave owns its 16-row strip
        __syncthreads();

        // LayerNorm over 128 cols; 2 threads per row; vectorized 128-bit LDS accesses
        {
          const float* g  = G  + (m*2 + l)*HID;
          const float* be = BE + (m*2 + l)*HID;
          const int r = t >> 1, half = t & 1;
          _Float16* hp = H + r*HSTR + half*64;    // 16B aligned (272B stride, 128B half)
          float s0 = 0.f, q0 = 0.f;
#pragma unroll
          for (int j = 0; j < 8; ++j) {
            v8h hv = *(const v8h*)(hp + j*8);
#pragma unroll
            for (int i = 0; i < 8; ++i) { float f = (float)hv[i]; s0 += f; q0 += f*f; }
          }
          REDS[t] = s0; REDQ[t] = q0;
          __syncthreads();
          float S  = REDS[r*2] + REDS[r*2 + 1];
          float Q  = REDQ[r*2] + REDQ[r*2 + 1];
          float mu = S * (1.f/128.f);
          float var = Q * (1.f/128.f) - mu*mu;
          float rs = rsqrtf(var + LN_EPS);
#pragma unroll
          for (int j = 0; j < 8; ++j) {
            v8h hv = *(const v8h*)(hp + j*8);
            v8h ov;
#pragma unroll
            for (int i = 0; i < 8; ++i) {
              int c = half*64 + j*8 + i;
              float f = (float)hv[i];
              ov[i] = (_Float16)((f - mu)*rs*g[c] + be[c]);
            }
            *(v8h*)(hp + j*8) = ov;
          }
        }
        __syncthreads();
      }

      // ---- head: [128 x 128] @ [128 x od] dot products; vectorized 128-bit LDS reads ----
      {
        int o = t >> 7, r = t & 127;
        int od = (m == 1) ? 1 : 2;
        if (o < od) {
          const _Float16* hp = H   + r*HSTR;          // 16B aligned
          const _Float16* wp = WLT + (m*2 + o)*HID;   // 256B rows, 16B aligned
          float sacc = BL[m*2 + o];
#pragma unroll
          for (int j = 0; j < 16; ++j) {
            v8h hv = *(const v8h*)(hp + j*8);
            v8h wv8 = *(const v8h*)(wp + j*8);
#pragma unroll
            for (int k = 0; k < 8; ++k) sacc += (float)hv[k] * (float)wv8[k];
          }
          OUTS[(m*TILE_M + r)*2 + o] = sacc;
        }
      }
      __syncthreads();
    }

    // ---- SDE step combine (DT = 1, sqrt_dt = 1) ----
    if (t < TILE_M) {
      const int  r   = t;
      const int  row = rowbase + r;
      const long gi  = (long)s*BATCH + row;
      float x     = XC[r];
      float inten = softplusf(OUTS[(1*TILE_M + r)*2 + 0]);
      float occ   = (uni[gi] < inten) ? 1.f : 0.f;
      float j0    = OUTS[(2*TILE_M + r)*2 + 0];
      float j1    = OUTS[(2*TILE_M + r)*2 + 1];
      float jump  = (j0 + expf(j1) * jnz[gi]) * occ;
      float drift = OUTS[(0*TILE_M + r)*2 + 0];
      float dcf   = softplusf(OUTS[(0*TILE_M + r)*2 + 1]);
      float xn    = x + drift + dcf * dnz[gi] + jump;
      XC[r] = xn;
      out[(long)row*(HORIZON + 1) + s + 1] = xn;
      if (s + 1 < HORIZON) {                   // hide next step's noise latency
        __builtin_prefetch(&uni[gi + BATCH], 0, 1);
        __builtin_prefetch(&jnz[gi + BATCH], 0, 1);
        __builtin_prefetch(&dnz[gi + BATCH], 0, 1);
      }
    }
    __syncthreads();
  }
}

// ---------------- host launch ----------------
extern "C" void kernel_launch(void* const* d_in, const int* in_sizes, int n_in,
                              void* d_out, int out_size, void* d_ws, size_t ws_size,
                              hipStream_t stream) {
  (void)in_sizes; (void)n_in; (void)d_ws; (void)ws_size; (void)out_size;
  // params flattened with sorted dict keys: {diff, intensity, jump} x {Ws[0..3], be[0..1], bs[0..3], g[0..1]}
  Params P;
  for (int m = 0; m < 3; ++m) {
    const int base = m * 12;
    for (int i = 0; i < 4; ++i) P.mlp[m].W[i] = (const float*)d_in[base + i];
    P.mlp[m].e[0] = (const float*)d_in[base + 4];
    P.mlp[m].e[1] = (const float*)d_in[base + 5];
    for (int i = 0; i < 4; ++i) P.mlp[m].b[i] = (const float*)d_in[base + 6 + i];
    P.mlp[m].g[0] = (const float*)d_in[base + 10];
    P.mlp[m].g[1] = (const float*)d_in[base + 11];
  }
  const float* x0   = (const float*)d_in[36];
  const float* cond = (const float*)d_in[37];
  const float* uni  = (const float*)d_in[38];
  const float* jnz  = (const float*)d_in[39];
  const float* dnz  = (const float*)d_in[40];
  float* out = (float*)d_out;

  hipFuncSetAttribute(reinterpret_cast<const void*>(&neural_jump_sde_kernel),
                      hipFuncAttributeMaxDynamicSharedMemorySize, (int)SHMEM_BYTES);
  neural_jump_sde_kernel<<<dim3(BATCH / TILE_M), dim3(NTHREADS), SHMEM_BYTES, stream>>>(
      P, x0, cond, uni, jnz, dnz, out);
}
